// RNN_Model_74723841016276
// MI455X (gfx1250) — matrix operand order repaired and checked by reading, not compiled, not run on hardware
//
#include <hip/hip_runtime.h>
#include <hip/hip_bf16.h>

// ---------------------------------------------------------------------------
// Shapes
// ---------------------------------------------------------------------------
#define B_  64
#define S_  2048
#define I_  128
#define H_  256
#define O_  64
#define RS  264   // padded LDS row stride for h (bf16): 528B -> 4-bank rotation
#define RSX 136   // padded LDS row stride for x (bf16): 272B -> 4-bank rotation

typedef __attribute__((ext_vector_type(16))) __bf16 v16bf;
typedef __attribute__((ext_vector_type(8)))  __bf16 v8bf;
typedef __attribute__((ext_vector_type(8)))  float  v8f;

// ---------------------------------------------------------------------------
// WMMA wrapper: D = A(16x32 bf16) * B(32x16 bf16) + C(16x16 f32)
// ---------------------------------------------------------------------------
__device__ __forceinline__ v8f wmma_bf16(v16bf a, v16bf b, v8f c) {
  return __builtin_amdgcn_wmma_f32_16x16x32_bf16(
      /*neg_a=*/false, a, /*neg_b=*/false, b,
      /*c_mod=*/(short)0, c, /*reuse_a=*/false, /*reuse_b=*/false);
}

__device__ __forceinline__ v16bf join16(v8bf lo, v8bf hi) {
  union { v16bf v; v8bf h[2]; } u;
  u.h[0] = lo; u.h[1] = hi;
  return u.v;
}

// Fast tanh for bf16 storage: 1 - 2*rcp(exp(2x)+1); exact at +-inf,
// error far below bf16 rounding. v_exp_f32 + v_rcp_f32, no branches.
__device__ __forceinline__ float fast_tanh(float v) {
  float e = __expf(2.0f * v);
  return 1.0f - 2.0f * __builtin_amdgcn_rcpf(e + 1.0f);
}

// B-fragment load from prepacked global buffer: 32 contiguous bytes per lane.
__device__ __forceinline__ v16bf ldB(const __bf16* fr, int fi, int lane) {
  const __bf16* p = fr + (size_t)fi * 512 + lane * 16;
  v8bf lo = *(const v8bf*)p;
  v8bf hi = *(const v8bf*)(p + 8);
  return join16(lo, hi);
}

// A-fragment (16 rows) from LDS (row-major, padded rows, stride `stride`).
// A k order used here (contiguous): lane m=L%16; lanes<16 hold K 0..15, lanes>=16 hold K 16..31.
__device__ __forceinline__ v16bf ldA(const __bf16* hb, int kt, int lane, int stride) {
  int m    = lane & 15;
  int koff = (lane & 16) ? 16 : 0;
  const __bf16* p = hb + m * stride + kt * 32 + koff;
  v8bf lo = *(const v8bf*)p;
  v8bf hi = *(const v8bf*)(p + 8);
  return join16(lo, hi);
}

// Stage x_t slice ([16 rows][128 cols] bf16) from global into padded LDS tile.
// 256 chunks of 8 elems; waves 0-7 copy one chunk each (uniform per wave).
__device__ __forceinline__ void stage_x(const __bf16* __restrict__ xbf,
                                        __bf16* dst, int s, int brow0, int tid) {
  for (int i = tid; i < 256; i += 512) {
    int row = i >> 4;
    int col = (i & 15) << 3;
    *(v8bf*)(dst + row * RSX + col) =
        *(const v8bf*)(xbf + ((size_t)(brow0 + row) * S_ + s) * I_ + col);
  }
}

// ---------------------------------------------------------------------------
// Convert x to bf16 once (same [B][S][I] layout).
// ---------------------------------------------------------------------------
__global__ void pack_x(const float* __restrict__ x, __bf16* __restrict__ xbf) {
  size_t idx = ((size_t)blockIdx.x * 256 + threadIdx.x) * 4;
#pragma unroll
  for (int i = 0; i < 4; i++) xbf[idx + i] = (__bf16)x[idx + i];
}

// ---------------------------------------------------------------------------
// Pack recurrent weights into WMMA B fragments (bf16).
// B k order used here (contiguous, as ldA): lane n=L%16; lanes 0-15 K=0..15, lanes 16-31
// K=16..31; element e of v16bf <-> K = e (+16 for upper lanes).
// frag element index f = ((kt*16 + nt)*32 + L)*16 + e
// ---------------------------------------------------------------------------
__global__ void pack_weights(const float* U0, const float* W0,
                             const float* U1, const float* W1, __bf16* wfrag) {
  int idx = blockIdx.x * 256 + threadIdx.x;
  const float* src; int K, f, base;
  if      (idx <  32768) { src = U0; K = 128; f = idx;          base = 0;      }
  else if (idx <  98304) { src = W0; K = 256; f = idx -  32768; base = 32768;  }
  else if (idx < 163840) { src = U1; K = 256; f = idx -  98304; base = 98304;  }
  else if (idx < 229376) { src = W1; K = 256; f = idx - 163840; base = 163840; }
  else return;
  int e  = f & 15;
  int L  = (f >> 4) & 31;
  int nt = (f >> 9) & 15;
  int kt = f >> 13;
  int n  = nt * 16 + (L & 15);
  int k  = kt * 32 + ((L >= 16) ? 16 : 0) + e;
  // GEMM is h @ M^T  =>  B[k][n] = M[n][k]
  wfrag[base + f] = (__bf16)src[n * K + k];
}

// ---------------------------------------------------------------------------
// Fold the affine head: T1 = fc2W @ fc1W ; t1b = fc2W@fc1b + fc2b
// ---------------------------------------------------------------------------
__global__ void fold1(const float* fc1W, const float* fc1b,
                      const float* fc2W, const float* fc2b,
                      float* T1, float* t1b) {
  int idx = blockIdx.x * 256 + threadIdx.x;
  if (idx < 65536) {
    int a = idx >> 8, k = idx & 255;
    float s = 0.f;
    for (int j = 0; j < 256; j++) s += fc2W[a * 256 + j] * fc1W[j * 256 + k];
    T1[idx] = s;
  }
  if (idx < 256) {
    float s = fc2b[idx];
    for (int j = 0; j < 256; j++) s += fc2W[idx * 256 + j] * fc1b[j];
    t1b[idx] = s;
  }
}

// Weff = fc3W @ T1 (written straight into packed bf16 B fragments, N=64)
// beff = fc3W @ t1b + fc3b
__global__ void fold2(const float* fc3W, const float* fc3b,
                      const float* T1, const float* t1b,
                      __bf16* hfrag, float* beff) {
  int idx = blockIdx.x * 256 + threadIdx.x;   // 16384 = 8kt * 4nt * 32L * 16e
  if (idx < 16384) {
    int e  = idx & 15;
    int L  = (idx >> 4) & 31;
    int nt = (idx >> 9) & 3;
    int kt = idx >> 11;
    int o  = nt * 16 + (L & 15);
    int k  = kt * 32 + ((L >= 16) ? 16 : 0) + e;
    float s = 0.f;
    for (int j = 0; j < 256; j++) s += fc3W[o * 256 + j] * T1[j * 256 + k];
    hfrag[idx] = (__bf16)s;   // B[k][o] = Weff[o][k]
  }
  if (idx < 64) {
    float s = fc3b[idx];
    for (int j = 0; j < 256; j++) s += fc3W[idx * 256 + j] * t1b[j];
    beff[idx] = s;
  }
}

// ---------------------------------------------------------------------------
// Persistent recurrence + fused head, batch-parallel.
// Batch rows are independent chains -> 4 workgroups, each owns a 16-row batch
// slice (one WMMA M-tile) on its own WGP. 16 waves per workgroup; wave w owns
// hidden-column tile w for both layers. Two interleaved accumulator chains
// per GEMM break the serial WMMA D->C dependency. h0/h1 double-buffered in
// LDS (2 barriers/step); x_t staged into LDS with a lead-2 pipeline.
// ---------------------------------------------------------------------------
__global__ void __launch_bounds__(512, 1)
rnn_persistent(const __bf16* __restrict__ xbf,
               const float* bU0, const float* bW0,
               const float* bU1, const float* bW1,
               const __bf16* wfrag, const __bf16* hfrag, const float* beff,
               float* __restrict__ out) {
  __shared__ __bf16 h0b[2][16 * RS];
  __shared__ __bf16 h1b[2][16 * RS];
  __shared__ __bf16 xls[2][16 * RSX];

  const int tid   = threadIdx.x;
  const int lane  = tid & 31;
  const int w     = tid >> 5;           // 0..15: hidden-column tile
  const int brow0 = blockIdx.x * 16;    // batch slice base row

  for (int i = tid; i < 16 * RS; i += 512) {
    h0b[0][i] = (__bf16)0.0f; h0b[1][i] = (__bf16)0.0f;
    h1b[0][i] = (__bf16)0.0f; h1b[1][i] = (__bf16)0.0f;
  }
  stage_x(xbf, xls[0], 0, brow0, tid);
  stage_x(xbf, xls[1], 1, brow0, tid);
  __syncthreads();

  const __bf16* fU0 = wfrag;            // KT=4, NT=16
  const __bf16* fW0 = wfrag + 32768;    // KT=8
  const __bf16* fU1 = wfrag + 98304;    // KT=8
  const __bf16* fW1 = wfrag + 163840;   // KT=8

  const int n  = w * 16 + (lane & 15);  // this lane's hidden-unit column
  const float b0 = bU0[n] + bW0[n];
  const float b1 = bU1[n] + bW1[n];

  // head: [16 x 64] per workgroup = 4 N-tiles; waves 0..3 own one each
  const int hnt = w & 3;
  const int ho  = hnt * 16 + (lane & 15);
  const float hbias = beff[ho];
  v16bf hB[8];
#pragma unroll
  for (int kt = 0; kt < 8; kt++) hB[kt] = ldB(hfrag, kt * 4 + hnt, lane);

  const int mbase = (lane & 16) ? 8 : 0;
  int p = 0;                            // "old" h-state buffer index

  for (int s = 0; s < S_; s++) {
    const __bf16* xcur = xls[s & 1];

    // ---------------- layer 0: h0' = tanh(x_t U0^T + h0 W0^T + b0) ----------
    v8f aA, aB;
#pragma unroll
    for (int r = 0; r < 8; r++) { aA[r] = b0; aB[r] = 0.0f; }
#pragma unroll
    for (int kt = 0; kt < 4; kt++) {
      v16bf bf = ldB(fU0, kt * 16 + w, lane);
      v16bf xa = ldA(xcur, kt, lane, RSX);
      if (kt & 1) aB = wmma_bf16(xa, bf, aB);
      else        aA = wmma_bf16(xa, bf, aA);
    }
#pragma unroll
    for (int kt = 0; kt < 8; kt++) {
      v16bf bf = ldB(fW0, kt * 16 + w, lane);
      v16bf ha = ldA(h0b[p], kt, lane, RS);
      if (kt & 1) aB = wmma_bf16(ha, bf, aB);
      else        aA = wmma_bf16(ha, bf, aA);
    }
    {
      v8f acc = aA + aB;
      __bf16* dst = h0b[p ^ 1];
#pragma unroll
      for (int r = 0; r < 8; r++)
        dst[(mbase + r) * RS + n] = (__bf16)fast_tanh(acc[r]);
    }
    __syncthreads();   // barrier 1: h0' complete; x_t reads complete

    // Stage x[s+2] into the buffer step s just finished reading.
    // Overlaps the layer-1 matrix work below; visible to readers at step s+2
    // via barrier 2 of this step (+ both barriers of step s+1).
    if (s + 2 < S_) stage_x(xbf, xls[s & 1], s + 2, brow0, tid);

    // ---------------- layer 1: h1' = tanh(h0' U1^T + h1 W1^T + b1) ----------
#pragma unroll
    for (int r = 0; r < 8; r++) { aA[r] = b1; aB[r] = 0.0f; }
#pragma unroll
    for (int kt = 0; kt < 8; kt++) {
      v16bf bf = ldB(fU1, kt * 16 + w, lane);
      v16bf ha = ldA(h0b[p ^ 1], kt, lane, RS);
      if (kt & 1) aB = wmma_bf16(ha, bf, aB);
      else        aA = wmma_bf16(ha, bf, aA);
    }
#pragma unroll
    for (int kt = 0; kt < 8; kt++) {
      v16bf bf = ldB(fW1, kt * 16 + w, lane);
      v16bf ha = ldA(h1b[p], kt, lane, RS);
      if (kt & 1) aB = wmma_bf16(ha, bf, aB);
      else        aA = wmma_bf16(ha, bf, aA);
    }
    {
      v8f acc = aA + aB;
      __bf16* dst = h1b[p ^ 1];
#pragma unroll
      for (int r = 0; r < 8; r++)
        dst[(mbase + r) * RS + n] = (__bf16)fast_tanh(acc[r]);
    }
    __syncthreads();   // barrier 2: h1' complete

    // ---------------- fused head: y[s] = h1' @ Weff^T + beff ---------------
    if (w < 4) {       // wave-uniform branch; EXEC all-ones inside
      v8f yA, yB;
#pragma unroll
      for (int r = 0; r < 8; r++) { yA[r] = hbias; yB[r] = 0.0f; }
#pragma unroll
      for (int kt = 0; kt < 8; kt++) {
        v16bf ha = ldA(h1b[p ^ 1], kt, lane, RS);
        if (kt & 1) yB = wmma_bf16(ha, hB[kt], yB);
        else        yA = wmma_bf16(ha, hB[kt], yA);
      }
      v8f ya = yA + yB;
#pragma unroll
      for (int r = 0; r < 8; r++)
        out[((size_t)(brow0 + mbase + r) * S_ + s) * O_ + ho] = ya[r];
    }

    p ^= 1;
  }
}

// ---------------------------------------------------------------------------
// Launch
// ---------------------------------------------------------------------------
extern "C" void kernel_launch(void* const* d_in, const int* in_sizes, int n_in,
                              void* d_out, int out_size, void* d_ws, size_t ws_size,
                              hipStream_t stream) {
  const float* x    = (const float*)d_in[0];
  const float* U0   = (const float*)d_in[1];
  const float* bU0  = (const float*)d_in[2];
  const float* W0   = (const float*)d_in[3];
  const float* bW0  = (const float*)d_in[4];
  const float* U1   = (const float*)d_in[5];
  const float* bU1  = (const float*)d_in[6];
  const float* W1   = (const float*)d_in[7];
  const float* bW1  = (const float*)d_in[8];
  const float* fc1W = (const float*)d_in[9];
  const float* fc1b = (const float*)d_in[10];
  const float* fc2W = (const float*)d_in[11];
  const float* fc2b = (const float*)d_in[12];
  const float* fc3W = (const float*)d_in[13];
  const float* fc3b = (const float*)d_in[14];
  float* out = (float*)d_out;

  char* ws = (char*)d_ws;
  __bf16* xbf   = (__bf16*)(ws);             // 16M bf16 = 33,554,432 B
  __bf16* wfrag = (__bf16*)(ws + 33554432);  // 229376 bf16 = 458,752 B
  float*  T1    = (float*)(ws + 34013184);   // 256x256 f32 = 262,144 B
  float*  t1b   = (float*)(ws + 34275328);   // 256 f32
  __bf16* hfrag = (__bf16*)(ws + 34276352);  // 16384 bf16 = 32,768 B
  float*  beff  = (float*)(ws + 34309120);   // 64 f32

  pack_x<<<16384, 256, 0, stream>>>(x, xbf);
  pack_weights<<<896, 256, 0, stream>>>(U0, W0, U1, W1, wfrag);
  fold1<<<256, 256, 0, stream>>>(fc1W, fc1b, fc2W, fc2b, T1, t1b);
  fold2<<<64, 256, 0, stream>>>(fc3W, fc3b, T1, t1b, hfrag, beff);
  rnn_persistent<<<4, 512, 0, stream>>>(xbf, bU0, bW0, bU1, bW1,
                                        wfrag, hfrag, beff, out);
}
